// GNNStack_26551487824702
// MI455X (gfx1250) — matrix operand order repaired and checked by reading, not compiled, not run on hardware
//
#include <hip/hip_runtime.h>

#define D_FEAT 256
#define HHEADS 4
#define CCH    64
#define LLAYERS 3
#define GG     64
#define OUTC   10
#define NEG_SLOPE 0.2f

typedef __bf16 bf16_t;
typedef __attribute__((ext_vector_type(16))) __bf16 v16bf;
typedef __attribute__((ext_vector_type(8)))  __bf16 v8bf;
typedef __attribute__((ext_vector_type(8)))  float  v8f;

// ---------- helpers ----------
__device__ __forceinline__ unsigned f2key(float x) {
    unsigned b = __float_as_uint(x);
    return (b & 0x80000000u) ? ~b : (b | 0x80000000u);
}
__device__ __forceinline__ float key2f(unsigned k) {
    unsigned b = (k & 0x80000000u) ? (k ^ 0x80000000u) : ~k;
    return __uint_as_float(b);
}
__device__ __forceinline__ void edge_nodes(const int* __restrict__ ei, int Ee,
                                           int e, int& s, int& d) {
    if (e < Ee) { s = ei[e]; d = ei[Ee + e]; }
    else        { s = e - Ee; d = e - Ee; }          // self-loop
}

// ---------- small utility kernels ----------
__global__ void k_cvt_bf16(const float* __restrict__ in, bf16_t* __restrict__ out, int n) {
    int i = blockIdx.x * blockDim.x + threadIdx.x;
    if (i < n) out[i] = (bf16_t)in[i];
}
__global__ void k_zero_f32(float* p, int n) {
    int i = blockIdx.x * blockDim.x + threadIdx.x;
    if (i < n) p[i] = 0.0f;
}
__global__ void k_init_mkey(unsigned* p, int n) {
    int i = blockIdx.x * blockDim.x + threadIdx.x;
    if (i < n) p[i] = f2key(-__builtin_inff());
}

// ---------- bf16 WMMA GEMM:  out[N,256] = h[N,256] @ W[256,256]^T ----------
// one wave -> one 16x16 f32 tile, K swept in steps of 32 (8x v_wmma_f32_16x16x32_bf16)
__global__ void __launch_bounds__(128)
k_gemm(const bf16_t* __restrict__ hbf,
       const bf16_t* __restrict__ Wsrc, const bf16_t* __restrict__ Wdst,
       float* __restrict__ xs, float* __restrict__ xd, int Nn)
{
    const bf16_t* W  = blockIdx.z ? Wdst : Wsrc;
    float*       out = blockIdx.z ? xd   : xs;

    const int wave = threadIdx.x >> 5;
    const int lane = threadIdx.x & 31;
    const int lm   = lane & 15;
    const int lh   = lane >> 4;

    const int row0 = blockIdx.x << 4;
    const int col0 = (blockIdx.y << 6) + (wave << 4);

    int arow = row0 + lm; if (arow >= Nn) arow = Nn - 1;   // clamp (N%16==0 normally)
    const bf16_t* aptr = hbf + (size_t)arow * D_FEAT + lh * 16;  // A: 16x32 bf16 layout, k = 16h + s
    const bf16_t* bptr = W   + (size_t)(col0 + lm) * D_FEAT + lh * 16; // B: 32x16 layout

    v8f acc = {};
#pragma unroll
    for (int k = 0; k < D_FEAT; k += 32) {
        v8bf alo = *(const v8bf*)(aptr + k);
        v8bf ahi = *(const v8bf*)(aptr + k + 8);
        v16bf a;
#pragma unroll
        for (int i = 0; i < 8; ++i) { a[i] = alo[i]; a[i + 8] = ahi[i]; }
        v16bf b = *(const v16bf*)(bptr + k);
        acc = __builtin_amdgcn_wmma_f32_16x16x32_bf16(
                  false, a, false, b, (short)0, acc, false, false);
    }

    const int orow = row0 + lh * 8;   // D: vgpr r -> row (lane/16)*8 + r, col lane%16
#pragma unroll
    for (int r = 0; r < 8; ++r) {
        int rr = orow + r;
        if (rr < Nn) out[(size_t)rr * D_FEAT + col0 + lm] = acc[r];
    }
}

// ---------- per-node attention logits ----------
__global__ void k_attn(const float* __restrict__ xs, const float* __restrict__ xd,
                       const float* __restrict__ aS, const float* __restrict__ aD,
                       float* __restrict__ als, float* __restrict__ ald, int NH)
{
    int i = blockIdx.x * blockDim.x + threadIdx.x;
    if (i >= NH) return;
    int n = i >> 2, h = i & 3;
    const float* ps = xs + (size_t)n * D_FEAT + h * CCH;
    const float* pd = xd + (size_t)n * D_FEAT + h * CCH;
    const float* vs = aS + h * CCH;
    const float* vd = aD + h * CCH;
    float ss = 0.f, sd = 0.f;
#pragma unroll 8
    for (int c = 0; c < CCH; ++c) { ss += ps[c] * vs[c]; sd += pd[c] * vd[c]; }
    als[i] = ss; ald[i] = sd;
}

// ---------- edge logit + segment max ----------
__global__ void k_edge_max(const int* __restrict__ ei, int Ee, int Et,
                           const float* __restrict__ als, const float* __restrict__ ald,
                           float* __restrict__ elog, unsigned* __restrict__ mkey)
{
    int idx = blockIdx.x * blockDim.x + threadIdx.x;
    if (idx >= Et * HHEADS) return;
    int e = idx >> 2, h = idx & 3;
    int s, d; edge_nodes(ei, Ee, e, s, d);
    float t = als[s * HHEADS + h] + ald[d * HHEADS + h];
    float v = t > 0.f ? t : NEG_SLOPE * t;
    elog[idx] = v;
    atomicMax(&mkey[d * HHEADS + h], f2key(v));
}

// ---------- exp(e - max) + segment sum ----------
__global__ void k_edge_exp(const int* __restrict__ ei, int Ee, int Et,
                           float* __restrict__ elog, const unsigned* __restrict__ mkey,
                           float* __restrict__ denom)
{
    int idx = blockIdx.x * blockDim.x + threadIdx.x;
    if (idx >= Et * HHEADS) return;
    int e = idx >> 2, h = idx & 3;
    int s, d; edge_nodes(ei, Ee, e, s, d);
    float w = __expf(elog[idx] - key2f(mkey[d * HHEADS + h]));
    elog[idx] = w;
    atomicAdd(&denom[d * HHEADS + h], w);
}

// ---------- normalize per-edge weight ----------
__global__ void k_edge_norm(const int* __restrict__ ei, int Ee, int Et,
                            float* __restrict__ elog, const float* __restrict__ denom)
{
    int idx = blockIdx.x * blockDim.x + threadIdx.x;
    if (idx >= Et * HHEADS) return;
    int e = idx >> 2, h = idx & 3;
    int s, d; edge_nodes(ei, Ee, e, s, d);
    elog[idx] = elog[idx] / denom[d * HHEADS + h];
}

// ---------- weighted message scatter-add ----------
__global__ void k_scatter(const int* __restrict__ ei, int Ee, int Et,
                          const float* __restrict__ xs, const float* __restrict__ wn,
                          float* __restrict__ agg)
{
    int idx = blockIdx.x * blockDim.x + threadIdx.x;
    if (idx >= Et * D_FEAT) return;
    int e = idx >> 8, c = idx & 255, h = c >> 6;
    int s, d; edge_nodes(ei, Ee, e, s, d);
    float w = wn[e * HHEADS + h];
    atomicAdd(&agg[(size_t)d * D_FEAT + c], xs[(size_t)s * D_FEAT + c] * w);
}

// ---------- bias + (optional) relu ----------
__global__ void k_epilogue(const float* __restrict__ agg, const float* __restrict__ bias_l,
                           float* __restrict__ hout, int n, int do_relu)
{
    int i = blockIdx.x * blockDim.x + threadIdx.x;
    if (i >= n) return;
    float v = agg[i] + bias_l[i & (D_FEAT - 1)];
    hout[i] = (do_relu && v < 0.f) ? 0.f : v;
}

// ---------- mean pool ----------
__global__ void k_pool(const float* __restrict__ h, const int* __restrict__ batch,
                       float* __restrict__ pooled, float* __restrict__ cnt, int Nn)
{
    int idx = blockIdx.x * blockDim.x + threadIdx.x;
    if (idx >= Nn * D_FEAT) return;
    int n = idx >> 8, c = idx & 255;
    int g = batch[n];
    atomicAdd(&pooled[g * D_FEAT + c], h[idx]);
    if (c == 0) atomicAdd(&cnt[g], 1.0f);
}

// ---------- classifier head ----------
__global__ void k_final(const float* __restrict__ pooled, const float* __restrict__ cnt,
                        const float* __restrict__ Wout, const float* __restrict__ bout,
                        float* __restrict__ out)
{
    int t = blockIdx.x * blockDim.x + threadIdx.x;
    if (t >= GG * OUTC) return;
    int g = t / OUTC, o = t % OUTC;
    float c = cnt[g]; if (c < 1.f) c = 1.f;
    float acc = 0.f;
    for (int k = 0; k < D_FEAT; ++k) acc += pooled[g * D_FEAT + k] * Wout[o * D_FEAT + k];
    out[t] = acc / c + bout[o];
}

// =====================================================================
extern "C" void kernel_launch(void* const* d_in, const int* in_sizes, int n_in,
                              void* d_out, int out_size, void* d_ws, size_t ws_size,
                              hipStream_t stream)
{
    const float* x     = (const float*)d_in[0];
    const int*   ei    = (const int*)  d_in[1];
    const int*   batch = (const int*)  d_in[2];
    const float* Wsrc  = (const float*)d_in[4];
    const float* Wdst  = (const float*)d_in[5];
    const float* attS  = (const float*)d_in[6];
    const float* attD  = (const float*)d_in[7];
    const float* bias  = (const float*)d_in[8];
    const float* Wout  = (const float*)d_in[9];
    const float* bout  = (const float*)d_in[10];
    float* out = (float*)d_out;

    const int Nn = in_sizes[0] / D_FEAT;       // 30000
    const int Ee = in_sizes[1] / 2;            // 480000
    const int Et = Ee + Nn;                    // + self loops
    const int ND = Nn * D_FEAT;
    const int NH = Nn * HHEADS;
    const int WSZ = D_FEAT * D_FEAT;           // 65536 per layer-matrix

    // ---- bump allocator over d_ws ----
    char* p = (char*)d_ws;
    auto alloc = [&](size_t bytes) -> void* {
        void* r = (void*)p;
        p += (bytes + 255) & ~(size_t)255;
        return r;
    };
    float*    hbuf  = (float*)   alloc((size_t)ND * 4);
    float*    xs    = (float*)   alloc((size_t)ND * 4);
    float*    xd    = (float*)   alloc((size_t)ND * 4);   // reused as agg
    bf16_t*   hbf   = (bf16_t*)  alloc((size_t)ND * 2);
    bf16_t*   wbfS  = (bf16_t*)  alloc((size_t)LLAYERS * WSZ * 2);
    bf16_t*   wbfD  = (bf16_t*)  alloc((size_t)LLAYERS * WSZ * 2);
    float*    als   = (float*)   alloc((size_t)NH * 4);
    float*    ald   = (float*)   alloc((size_t)NH * 4);
    unsigned* mkey  = (unsigned*)alloc((size_t)NH * 4);
    float*    denom = (float*)   alloc((size_t)NH * 4);
    float*    wnorm = (float*)   alloc((size_t)Et * HHEADS * 4);
    float*    pooled= (float*)   alloc((size_t)GG * D_FEAT * 4);
    float*    cnt   = (float*)   alloc((size_t)GG * 4);

    const int T = 256;
    auto nb = [](int work, int thr) { return (work + thr - 1) / thr; };

    // convert all weights to bf16 once
    k_cvt_bf16<<<nb(LLAYERS * WSZ, T), T, 0, stream>>>(Wsrc, wbfS, LLAYERS * WSZ);
    k_cvt_bf16<<<nb(LLAYERS * WSZ, T), T, 0, stream>>>(Wdst, wbfD, LLAYERS * WSZ);

    const int rowTiles = (Nn + 15) / 16;
    dim3 ggrid(rowTiles, D_FEAT / 64, 2), gblk(128, 1, 1);

    for (int l = 0; l < LLAYERS; ++l) {
        const float* h_in = (l == 0) ? x : hbuf;

        // 1) h -> bf16
        k_cvt_bf16<<<nb(ND, T), T, 0, stream>>>(h_in, hbf, ND);

        // 2) xs = h @ Wsrc^T ; xd = h @ Wdst^T   (WMMA bf16)
        k_gemm<<<ggrid, gblk, 0, stream>>>(hbf, wbfS + (size_t)l * WSZ,
                                           wbfD + (size_t)l * WSZ, xs, xd, Nn);

        // 3) per-node logits
        k_attn<<<nb(NH, T), T, 0, stream>>>(xs, xd,
                                            attS + l * HHEADS * CCH,
                                            attD + l * HHEADS * CCH,
                                            als, ald, NH);

        // 4) segment softmax over incoming edges
        k_init_mkey<<<nb(NH, T), T, 0, stream>>>(mkey, NH);
        k_zero_f32 <<<nb(NH, T), T, 0, stream>>>(denom, NH);
        k_edge_max <<<nb(Et * HHEADS, T), T, 0, stream>>>(ei, Ee, Et, als, ald, wnorm, mkey);
        k_edge_exp <<<nb(Et * HHEADS, T), T, 0, stream>>>(ei, Ee, Et, wnorm, mkey, denom);
        k_edge_norm<<<nb(Et * HHEADS, T), T, 0, stream>>>(ei, Ee, Et, wnorm, denom);

        // 5) weighted scatter-add (xd buffer reused as aggregator after logits done)
        float* agg = xd;
        k_zero_f32<<<nb(ND, T), T, 0, stream>>>(agg, ND);
        k_scatter <<<nb(Et * D_FEAT, T), T, 0, stream>>>(ei, Ee, Et, xs, wnorm, agg);

        // 6) bias + relu (except last layer)
        k_epilogue<<<nb(ND, T), T, 0, stream>>>(agg, bias + l * D_FEAT, hbuf, ND,
                                                (l < LLAYERS - 1) ? 1 : 0);
    }

    // mean pool + classifier
    k_zero_f32<<<nb(GG * D_FEAT, T), T, 0, stream>>>(pooled, GG * D_FEAT);
    k_zero_f32<<<nb(GG, T), T, 0, stream>>>(cnt, GG);
    k_pool <<<nb(ND, T), T, 0, stream>>>(hbuf, batch, pooled, cnt, Nn);
    k_final<<<nb(GG * OUTC, T), T, 0, stream>>>(pooled, cnt, Wout, bout, out);
}
